// Multifaceted_Attention_Mixer_2714419331127
// MI455X (gfx1250) — compile-verified, matrix operand order checked by reading, not run on hardware
//
#include <hip/hip_runtime.h>
#include <hip/hip_bf16.h>

// ---------------- problem constants ----------------
#define B_   512
#define S_   200
#define D_   512
#define H_   8
#define DH_  64
#define BS_  (B_ * S_)          // 102400
#define LN_EPS 1e-12f

typedef __attribute__((ext_vector_type(16))) _Float16 v16h;
typedef __attribute__((ext_vector_type(8)))  float    v8f;

// ---------------- CDNA5 async global->LDS copy ----------------
#define USE_ASYNC_LDS 1

__device__ __forceinline__ void async_b128(void* lds_dst, const void* gsrc) {
#if USE_ASYNC_LDS
  // generic pointer to shared memory: low 32 bits are the LDS byte offset
  unsigned ldsoff = (unsigned)(unsigned long long)(uintptr_t)lds_dst;
  asm volatile("global_load_async_to_lds_b128 %0, %1, off"
               :
               : "v"(ldsoff), "v"((unsigned long long)(uintptr_t)gsrc)
               : "memory");
#else
  *(int4*)lds_dst = *(const int4*)gsrc;
#endif
}

__device__ __forceinline__ void async_wait_all() {
#if USE_ASYNC_LDS
  asm volatile("s_wait_asynccnt 0x0" ::: "memory");
#endif
}

// ---------------- WMMA helpers ----------------
__device__ __forceinline__ v8f wmma16(v16h a, v16h b, v8f c) {
  // D = A(16x32 f16) * B(32x16 f16) + C(16x16 f32)
  return __builtin_amdgcn_wmma_f32_16x16x32_f16(
      /*neg_a=*/false, a, /*neg_b=*/false, b,
      /*c_mod=*/(short)0, c, /*reuse_a=*/false, /*reuse_b=*/false);
}

// element e of an A/B fragment maps to K = ((e<8)?e:e+8) + 8*(lane>>4)
// row-major over the non-K dim: f[e] = base[(lane&15)*stride + k0 + kmap(e)]
__device__ __forceinline__ v16h frag_rowmajor(const _Float16* base, int stride,
                                              int lane, int k0, int kmax) {
  const _Float16* r = base + (size_t)(lane & 15) * stride;
  const int half = lane >> 4;
  v16h f;
#pragma unroll
  for (int e = 0; e < 16; ++e) {
    int k = k0 + ((e < 8) ? e : e + 8) + 8 * half;
    f[e] = (k < kmax) ? r[k] : (_Float16)0.f;
  }
  return f;
}

// A-fragment with row clamping (for M beyond S_), unmasked K
__device__ __forceinline__ v16h frag_rowclamp(const _Float16* mat, int stride,
                                              int lane, int m0, int mmax, int k0) {
  int m = m0 + (lane & 15);
  if (m >= mmax) m = mmax - 1;
  const _Float16* r = mat + (size_t)m * stride;
  const int half = lane >> 4;
  v16h f;
#pragma unroll
  for (int e = 0; e < 16; ++e) {
    int k = k0 + ((e < 8) ? e : e + 8) + 8 * half;
    f[e] = r[k];
  }
  return f;
}

// B fragment from K-major storage ([k][n], row stride = nstride):
// lane -> n = lane&15 (caller pre-offsets to the tile's n base and k0 row)
__device__ __forceinline__ v16h fragB_kmajor(const _Float16* bp, int nstride,
                                             int lane, int krem) {
  const int n = lane & 15, half = lane >> 4;
  v16h f;
#pragma unroll
  for (int e = 0; e < 16; ++e) {
    int k = ((e < 8) ? e : e + 8) + 8 * half;
    f[e] = (k < krem) ? bp[(size_t)k * nstride + n] : (_Float16)0.f;
  }
  return f;
}

// ============================================================
// Kernel 0: pre-transpose weights to f16 [n][k] (one-time, 2MB total)
// grid = (8, 8, 4), block = 256 ; LDS-tiled so both sides are coalesced
// ============================================================
__global__ __launch_bounds__(256) void k_wtrans(
    const float* __restrict__ Wq, const float* __restrict__ Wk,
    const float* __restrict__ Wv, const float* __restrict__ Wd,
    _Float16* __restrict__ WT) {
  __shared__ float tile[64][65];
  const int mat = blockIdx.z;
  const float* W = (mat == 0) ? Wq : (mat == 1) ? Wk : (mat == 2) ? Wv : Wd;
  _Float16* Wt = WT + (size_t)mat * D_ * D_;
  const int k0 = blockIdx.x * 64, n0 = blockIdx.y * 64;
  const int c = threadIdx.x & 63, r0 = (threadIdx.x >> 6) * 16;
#pragma unroll
  for (int j = 0; j < 16; ++j) {
    const int r = r0 + j;
    tile[r][c] = W[(size_t)(k0 + r) * D_ + n0 + c];
  }
  __syncthreads();
#pragma unroll
  for (int j = 0; j < 16; ++j) {
    const int r = r0 + j;
    Wt[(size_t)(n0 + r) * D_ + k0 + c] = (_Float16)tile[c][r];
  }
}

// ============================================================
// Kernel 1: x = in + pos ; h = LayerNorm(x) -> h32 (fp32), h16 (f16)
// one wave per row, 8 waves / block
// ============================================================
__global__ __launch_bounds__(256) void k_fuse_ln(
    const float* __restrict__ x, const float* __restrict__ pe,
    const float* __restrict__ g, const float* __restrict__ bb,
    float* __restrict__ h32, _Float16* __restrict__ h16) {
  const int row  = blockIdx.x * 8 + (threadIdx.x >> 5);
  const int lane = threadIdx.x & 31;
  const float* xr = x  + (size_t)row * D_;
  const float* pr = pe + (size_t)row * D_;
  float v[16];
  float s1 = 0.f, s2 = 0.f;
#pragma unroll
  for (int i = 0; i < 16; ++i) {
    float t = xr[lane + i * 32] + pr[lane + i * 32];
    v[i] = t; s1 += t; s2 += t * t;
  }
#pragma unroll
  for (int off = 16; off >= 1; off >>= 1) {
    s1 += __shfl_xor(s1, off, 32);
    s2 += __shfl_xor(s2, off, 32);
  }
  const float mu  = s1 * (1.f / D_);
  const float var = s2 * (1.f / D_) - mu * mu;
  const float inv = rsqrtf(var + LN_EPS);
  float*     hr = h32 + (size_t)row * D_;
  _Float16*  hh = h16 + (size_t)row * D_;
#pragma unroll
  for (int i = 0; i < 16; ++i) {
    const int n = lane + i * 32;
    const float y = (v[i] - mu) * inv * g[n] + bb[n];
    hr[n] = y;
    hh[n] = (_Float16)y;
  }
}

// ============================================================
// Kernel 2: fused QKV projection + bias + ELU + per-head L2 norm
// grid = (BS/64, H, 3) ; block = 256 (8 waves)
// tile: 64 rows x 64 cols (one head), K loop over 512 in 32-chunks,
// double-buffered LDS fed entirely by async b128 DMA.
// z==0 -> qn [B,H,S,DH] ; z==1 -> knT [B,H,DH,S] ; z==2 -> v [B,H,S,DH]
// ============================================================
__global__ __launch_bounds__(256) void k_qkv(
    const _Float16* __restrict__ h16, const _Float16* __restrict__ WT,
    const float* __restrict__ bq, const float* __restrict__ bk,
    const float* __restrict__ bv,
    _Float16* __restrict__ qn, _Float16* __restrict__ knT,
    _Float16* __restrict__ vv) {
  const int zw = blockIdx.z;
  const _Float16* Wt  = WT + (size_t)zw * D_ * D_;  // [n][k] f16
  const float*    bias = (zw == 0) ? bq : (zw == 1) ? bk : bv;
  const int head = blockIdx.y;
  const int row0 = blockIdx.x * 64;
  const int n0   = head * DH_;

  __shared__ _Float16 As[2][64 * 32];   // [m][k]
  __shared__ _Float16 Bs[2][64 * 32];   // [n][k]
  __shared__ float    Xs[64 * 64];
  __shared__ float    psum[64][4];

  const int tid = threadIdx.x;
  const int wave = tid >> 5, lane = tid & 31;
  const int t0 = wave * 2;
  const int mt = t0 >> 2;
  const int ntA = t0 & 3, ntB = ntA + 1;

  auto load_chunk = [&](int kc, int buf) {
    const int k0 = kc * 32;
    const int r = tid >> 2, c = tid & 3;
    async_b128(&As[buf][r * 32 + c * 8],
               h16 + (size_t)(row0 + r) * D_ + k0 + c * 8);
    async_b128(&Bs[buf][r * 32 + c * 8],
               Wt + (size_t)(n0 + r) * D_ + k0 + c * 8);
  };

  v8f acc0 = {}; v8f acc1 = {};

  load_chunk(0, 0);
  async_wait_all();
  __syncthreads();

  for (int kc = 0; kc < 16; ++kc) {
    const int cur = kc & 1;
    if (kc + 1 < 16) load_chunk(kc + 1, cur ^ 1);
    if (kc + 2 < 16)  // keep next-next weight panel warm in L2/WGP$
      __builtin_prefetch(Wt + (size_t)(n0 + (tid & 63)) * D_ + (kc + 2) * 32, 0, 1);

    const v16h a  = frag_rowmajor(&As[cur][mt  * 16 * 32], 32, lane, 0, 1 << 30);
    const v16h b0 = frag_rowmajor(&Bs[cur][ntA * 16 * 32], 32, lane, 0, 1 << 30);
    acc0 = wmma16(a, b0, acc0);
    const v16h b1 = frag_rowmajor(&Bs[cur][ntB * 16 * 32], 32, lane, 0, 1 << 30);
    acc1 = wmma16(a, b1, acc1);

    async_wait_all();
    __syncthreads();
  }

  { // spill D tiles (+bias) into Xs
    const int nA = ntA * 16 + (lane & 15);
    const int nB = ntB * 16 + (lane & 15);
    const float biasA = bias[n0 + nA];
    const float biasB = bias[n0 + nB];
#pragma unroll
    for (int r = 0; r < 8; ++r) {
      const int m = mt * 16 + r + 8 * (lane >> 4);
      Xs[m * 64 + nA] = acc0[r] + biasA;
      Xs[m * 64 + nB] = acc1[r] + biasB;
    }
  }
  __syncthreads();

  // post-process: 4 threads per row, 16 cols each
  const int r  = tid >> 2, qq = tid & 3, c0 = qq * 16;
  const int grow = row0 + r;
  const int bidx = grow / S_, srow = grow % S_;
  float vals[16];
#pragma unroll
  for (int i = 0; i < 16; ++i) vals[i] = Xs[r * 64 + c0 + i];

  if (zw < 2) {  // ELU + per-row L2 norm over DH
    float ss = 0.f;
#pragma unroll
    for (int i = 0; i < 16; ++i) {
      const float xv = vals[i];
      const float y  = xv > 0.f ? xv : (__expf(xv) - 1.f);
      vals[i] = y; ss += y * y;
    }
    psum[r][qq] = ss;
    __syncthreads();
    const float tot = psum[r][0] + psum[r][1] + psum[r][2] + psum[r][3];
    const float rn  = rsqrtf(tot);
    if (zw == 0) {
      _Float16* dst = qn + (((size_t)bidx * H_ + head) * S_ + srow) * DH_ + c0;
#pragma unroll
      for (int i = 0; i < 16; ++i) dst[i] = (_Float16)(vals[i] * rn);
    } else {  // store K transposed: [B,H,DH,S]
      _Float16* dst = knT + ((size_t)bidx * H_ + head) * DH_ * S_ + srow;
#pragma unroll
      for (int i = 0; i < 16; ++i)
        dst[(size_t)(c0 + i) * S_] = (_Float16)(vals[i] * rn);
    }
  } else {  // V: raw
    _Float16* dst = vv + (((size_t)bidx * H_ + head) * S_ + srow) * DH_ + c0;
#pragma unroll
    for (int i = 0; i < 16; ++i) dst[i] = (_Float16)vals[i];
  }
}

// ============================================================
// Kernel 3: kv[b,h] = knT(64x200) @ v(200x64) ; K tail zero-masked
// grid = B*H ; block = 512 (16 waves, one 16x16 tile each)
// ============================================================
__global__ __launch_bounds__(512) void k_kv(
    const _Float16* __restrict__ knT, const _Float16* __restrict__ vv,
    _Float16* __restrict__ kv) {
  const int bh = blockIdx.x;
  const _Float16* A  = knT + (size_t)bh * DH_ * S_;   // [64][200]
  const _Float16* Bv = vv  + (size_t)bh * S_  * DH_;  // [200][64]
  const int tid = threadIdx.x, wave = tid >> 5, lane = tid & 31;
  const int mt = wave >> 2, nt = wave & 3;

  v8f acc = {};
#pragma unroll
  for (int kc = 0; kc < 7; ++kc) {
    const int k0 = kc * 32;
    const v16h a = frag_rowmajor(A + (size_t)(mt * 16) * S_, S_, lane, k0, S_);
    const v16h b = fragB_kmajor(Bv + (size_t)k0 * DH_ + nt * 16, DH_, lane, S_ - k0);
    acc = wmma16(a, b, acc);
  }
  const int n = nt * 16 + (lane & 15);
#pragma unroll
  for (int r = 0; r < 8; ++r) {
    const int m = mt * 16 + r + 8 * (lane >> 4);
    kv[((size_t)bh * DH_ + m) * DH_ + n] = (_Float16)acc[r];
  }
}

// ============================================================
// Kernel 4: ctx[b,h] = qn(200x64) @ kv(64x64) * (1/sqrt(64))
// grid = (ceil(S/16)=13, B*H) ; block = 128 (4 waves = 4 N-tiles)
// ============================================================
__global__ __launch_bounds__(128) void k_ctx(
    const _Float16* __restrict__ qn, const _Float16* __restrict__ kv,
    _Float16* __restrict__ ctx) {
  const int bh = blockIdx.y;
  const int bb_ = bh >> 3, head = bh & 7;
  const int m0 = blockIdx.x * 16;
  const int wave = threadIdx.x >> 5, lane = threadIdx.x & 31;
  const _Float16* A  = qn + (size_t)bh * S_  * DH_;   // [200][64]
  const _Float16* Bm = kv + (size_t)bh * DH_ * DH_;   // [64][64]

  v8f acc = {};
#pragma unroll
  for (int kc = 0; kc < 2; ++kc) {
    const int k0 = kc * 32;
    const v16h a = frag_rowclamp(A, DH_, lane, m0, S_, k0);
    const v16h b = fragB_kmajor(Bm + (size_t)k0 * DH_ + wave * 16, DH_, lane, 64);
    acc = wmma16(a, b, acc);
  }
  const int n = wave * 16 + (lane & 15);
#pragma unroll
  for (int r = 0; r < 8; ++r) {
    const int m = m0 + r + 8 * (lane >> 4);
    if (m < S_)
      ctx[((size_t)bb_ * S_ + m) * D_ + head * DH_ + n] =
          (_Float16)(acc[r] * 0.125f);
  }
}

// ============================================================
// Kernel 5: out = LayerNorm(ctx @ Wd + bd + h)
// grid = BS/32 ; block = 256 (8 waves); tile M=32 x N=512, K over 512,
// double-buffered async LDS panels; Ts (fp32 staging) aliases As/Bs.
// ============================================================
__global__ __launch_bounds__(256) void k_out(
    const _Float16* __restrict__ ctx, const _Float16* __restrict__ WdT,
    const float* __restrict__ bd, const float* __restrict__ h32,
    const float* __restrict__ g, const float* __restrict__ bb,
    float* __restrict__ out) {
  // 68KB: As0 [0,2K) As1 [2K,4K) Bs0 [4K,36K) Bs1 [36K,68K); Ts aliases [0,64K)
  __shared__ char smem[2 * 2048 + 2 * 32768];
  __shared__ float red[32][8][2];

  const int row0 = blockIdx.x * 32;
  const int tid = threadIdx.x, wave = tid >> 5, lane = tid & 31;
  const int nbase = wave * 64;

  // runtime offset computation only -- no constant arrays of LDS pointers
  auto As_of = [&](int buf) -> _Float16* {
    return (_Float16*)(smem + buf * 2048);
  };
  auto Bs_of = [&](int buf) -> _Float16* {
    return (_Float16*)(smem + 4096 + buf * 32768);
  };

  auto load_chunk = [&](int kc, int buf) {
    const int k0 = kc * 32;
    _Float16* Asb = As_of(buf);
    _Float16* Bsb = Bs_of(buf);
    if (tid < 128) {  // A panel: 32x32 f16 = 128 * 16B
      const int r = tid >> 2, c = tid & 3;
      async_b128(&Asb[r * 32 + c * 8],
                 ctx + (size_t)(row0 + r) * D_ + k0 + c * 8);
    }
    // B panel: WdT[n][k0..k0+31], 512 rows x 64B = 2048 * 16B
#pragma unroll
    for (int j = 0; j < 8; ++j) {
      const int idx = tid + (j << 8);
      const int r = idx >> 2, c = idx & 3;
      async_b128(&Bsb[r * 32 + c * 8],
                 WdT + (size_t)r * D_ + k0 + c * 8);
    }
  };

  v8f acc[8];
#pragma unroll
  for (int i = 0; i < 8; ++i) acc[i] = (v8f){};

  load_chunk(0, 0);
  async_wait_all();
  __syncthreads();

  for (int kc = 0; kc < 16; ++kc) {
    const int cur = kc & 1;
    if (kc + 1 < 16) load_chunk(kc + 1, cur ^ 1);
    const _Float16* Asb = As_of(cur);
    const _Float16* Bsb = Bs_of(cur);
#pragma unroll
    for (int mt2 = 0; mt2 < 2; ++mt2) {
      const v16h a = frag_rowmajor(&Asb[mt2 * 16 * 32], 32, lane, 0, 1 << 30);
#pragma unroll
      for (int nt = 0; nt < 4; ++nt) {
        const v16h b = frag_rowmajor(&Bsb[(size_t)(nbase + nt * 16) * 32],
                                     32, lane, 0, 1 << 30);
        acc[mt2 * 4 + nt] = wmma16(a, b, acc[mt2 * 4 + nt]);
      }
    }
    async_wait_all();
    __syncthreads();
  }

  // stage: t = gemm + bias + residual h (fp32); Ts aliases dead As/Bs
  float* Ts = (float*)smem;
#pragma unroll
  for (int mt2 = 0; mt2 < 2; ++mt2) {
#pragma unroll
    for (int nt = 0; nt < 4; ++nt) {
      const int n = nbase + nt * 16 + (lane & 15);
      const float bias = bd[n];
#pragma unroll
      for (int r = 0; r < 8; ++r) {
        const int m = mt2 * 16 + r + 8 * (lane >> 4);
        Ts[m * 512 + n] =
            acc[mt2 * 4 + nt][r] + bias + h32[(size_t)(row0 + m) * D_ + n];
      }
    }
  }
  __syncthreads();

  // final LayerNorm: 8 threads per row, 64 cols each
  const int r = tid >> 3, seg = tid & 7, c0 = seg * 64;
  float s1 = 0.f, s2 = 0.f;
#pragma unroll 8
  for (int i = 0; i < 64; ++i) {
    const float xv = Ts[r * 512 + c0 + i];
    s1 += xv; s2 += xv * xv;
  }
  red[r][seg][0] = s1; red[r][seg][1] = s2;
  __syncthreads();
  float t1 = 0.f, t2 = 0.f;
#pragma unroll
  for (int j = 0; j < 8; ++j) { t1 += red[r][j][0]; t2 += red[r][j][1]; }
  const float mu  = t1 * (1.f / D_);
  const float var = t2 * (1.f / D_) - mu * mu;
  const float inv = rsqrtf(var + LN_EPS);
  float* dst = out + (size_t)(row0 + r) * D_;
#pragma unroll 8
  for (int i = 0; i < 64; ++i) {
    const int n = c0 + i;
    dst[n] = (Ts[r * 512 + n] - mu) * inv * g[n] + bb[n];
  }
}

// ============================================================
extern "C" void kernel_launch(void* const* d_in, const int* in_sizes, int n_in,
                              void* d_out, int out_size, void* d_ws, size_t ws_size,
                              hipStream_t stream) {
  const float* x   = (const float*)d_in[0];
  const float* pe  = (const float*)d_in[1];
  // d_in[2] attention_mask: unused by LinMHA
  const float* Wq  = (const float*)d_in[3];
  const float* bq  = (const float*)d_in[4];
  const float* Wk  = (const float*)d_in[5];
  const float* bk  = (const float*)d_in[6];
  const float* Wv  = (const float*)d_in[7];
  const float* bv  = (const float*)d_in[8];
  const float* Wd  = (const float*)d_in[9];
  const float* bd  = (const float*)d_in[10];
  const float* lng = (const float*)d_in[11];
  const float* lnb = (const float*)d_in[12];
  float* out = (float*)d_out;

  char* ws = (char*)d_ws;
  size_t off = 0;
  float*    h32 = (float*)(ws + off);    off += (size_t)BS_ * D_ * sizeof(float);
  _Float16* h16 = (_Float16*)(ws + off); off += (size_t)BS_ * D_ * sizeof(_Float16);
  _Float16* qn  = (_Float16*)(ws + off); off += (size_t)BS_ * D_ * sizeof(_Float16);
  _Float16* knT = (_Float16*)(ws + off); off += (size_t)BS_ * D_ * sizeof(_Float16);
  _Float16* vv  = (_Float16*)(ws + off); off += (size_t)BS_ * D_ * sizeof(_Float16);
  _Float16* kv  = (_Float16*)(ws + off); off += (size_t)B_ * H_ * DH_ * DH_ * sizeof(_Float16);
  _Float16* ctx = (_Float16*)(ws + off); off += (size_t)BS_ * D_ * sizeof(_Float16);
  _Float16* WT  = (_Float16*)(ws + off); off += (size_t)4 * D_ * D_ * sizeof(_Float16);

  k_wtrans<<<dim3(8, 8, 4), 256, 0, stream>>>(Wq, Wk, Wv, Wd, WT);
  k_fuse_ln<<<BS_ / 8, 256, 0, stream>>>(x, pe, lng, lnb, h32, h16);
  k_qkv<<<dim3(BS_ / 64, H_, 3), 256, 0, stream>>>(h16, WT, bq, bk, bv,
                                                   qn, knT, vv);
  k_kv<<<B_ * H_, 512, 0, stream>>>(knT, vv, kv);
  k_ctx<<<dim3((S_ + 15) / 16, B_ * H_), 128, 0, stream>>>(qn, kv, ctx);
  k_out<<<BS_ / 32, 256, 0, stream>>>(ctx, WT + (size_t)3 * D_ * D_, bd, h32,
                                      lng, lnb, out);
}
